// RNNWithMovingAttentionKV_54425825575662
// MI455X (gfx1250) — matrix operand order repaired and checked by reading, not compiled, not run on hardware
//
#include <hip/hip_runtime.h>
#include <math.h>

#define EMBED   256
#define NSTEP   128
#define BATCH   64
#define VOCAB   32000
#define MROWS   (BATCH * NSTEP)   /* 8192 */
#define NCHUNKS 200               /* 32000 / 160 */
#define CHUNK_N 160               /* 2 wave-groups * 80 cols */
#define MTILE   64                /* rows per block */

typedef unsigned short u16;
typedef unsigned int   u32;

typedef __attribute__((ext_vector_type(16))) __bf16 v16bf;
typedef __attribute__((ext_vector_type(8)))  float  v8f;

union V16 { v16bf v; uint4 q[2]; };

__device__ __forceinline__ u16 f2bf(float x) {
    u32 u = __float_as_uint(x);
    u32 r = u + 0x7FFFu + ((u >> 16) & 1u);   // round-to-nearest-even
    return (u16)(r >> 16);
}

// ---------------------------------------------------------------------------
// Kernel 1: 128-step RNN scan. One block (256 threads) per batch element.
// Thread e carries z[e]. Writes zs in f32 (for exact target dots) and bf16
// (GEMM A matrix), row index m = b*128 + t.
// ---------------------------------------------------------------------------
__global__ __launch_bounds__(256) void rnn_scan_kernel(
    const int* __restrict__ zi, const float* __restrict__ latent,
    const float* __restrict__ kv, float* __restrict__ zs_f32,
    u16* __restrict__ zs_bf16)
{
    __shared__ float s_sk[10 * EMBED];
    __shared__ float s_sv[10 * EMBED];
    __shared__ float s_red[256];
    __shared__ float s_red10[256 * 10];
    __shared__ float s_att[10];

    const int b = blockIdx.x;
    const int t = threadIdx.x;          // == embedding index e

    for (int i = t; i < 10 * EMBED; i += 256) {
        s_sk[i] = kv[i];
        s_sv[i] = kv[10 * EMBED + i];
    }

    float z = latent[(size_t)zi[b] * (2 * EMBED * 10) + t];  // z_init = first 256
    __syncthreads();

    for (int step = 0; step < NSTEP; ++step) {
        // ---- mean ----
        s_red[t] = z;
        __syncthreads();
        for (int st = 128; st > 0; st >>= 1) {
            if (t < st) s_red[t] += s_red[t + st];
            __syncthreads();
        }
        const float mu = s_red[0] * (1.0f / 256.0f);
        const float d  = z - mu;
        __syncthreads();
        // ---- unbiased variance ----
        s_red[t] = d * d;
        __syncthreads();
        for (int st = 128; st > 0; st >>= 1) {
            if (t < st) s_red[t] += s_red[t + st];
            __syncthreads();
        }
        const float stddev = sqrtf(s_red[0] * (1.0f / 255.0f));
        const float zn = z / (1e-5f + stddev) * 0.113f;
        __syncthreads();
        // ---- att logits: 10 dot products over 256 ----
        #pragma unroll
        for (int j = 0; j < 10; ++j) s_red10[t * 10 + j] = zn * s_sk[j * EMBED + t];
        __syncthreads();
        for (int st = 128; st > 0; st >>= 1) {
            if (t < st) {
                #pragma unroll
                for (int j = 0; j < 10; ++j)
                    s_red10[t * 10 + j] += s_red10[(t + st) * 10 + j];
            }
            __syncthreads();
        }
        if (t == 0) {
            float mx = s_red10[0];
            #pragma unroll
            for (int j = 1; j < 10; ++j) mx = fmaxf(mx, s_red10[j]);
            float se = 0.f;
            float ex[10];
            #pragma unroll
            for (int j = 0; j < 10; ++j) { ex[j] = expf(s_red10[j] - mx); se += ex[j]; }
            const float inv = 1.0f / se;
            #pragma unroll
            for (int j = 0; j < 10; ++j) s_att[j] = ex[j] * inv;
        }
        __syncthreads();
        // ---- emit = att @ sv ----
        float znew = 0.f;
        #pragma unroll
        for (int j = 0; j < 10; ++j) znew += s_att[j] * s_sv[j * EMBED + t];
        z = znew;
        const size_t idx = ((size_t)b * NSTEP + step) * EMBED + t;
        zs_f32[idx]  = z;
        zs_bf16[idx] = f2bf(z);
        __syncthreads();
    }
}

// ---------------------------------------------------------------------------
// Kernel 2: vocab_w f32 -> bf16, grid-stride.
// ---------------------------------------------------------------------------
__global__ __launch_bounds__(256) void cvt_bf16_kernel(
    const float* __restrict__ in, u16* __restrict__ out, int n)
{
    for (int i = blockIdx.x * blockDim.x + threadIdx.x; i < n;
         i += gridDim.x * blockDim.x)
        out[i] = f2bf(in[i]);
}

// ---------------------------------------------------------------------------
// Kernel 3: bf16 WMMA GEMM (M=8192, N=32000, K=256) fused with per-row
// partial logsumexp over 160-column chunks. Block = 8 waves arranged 4(M)x2(N);
// each wave owns a 16x80 tile (5 accumulators), K swept in 8 steps of 32.
// The 64x256 A tile is staged once into LDS via async global->LDS copies
// (ASYNCcnt) and read back as WMMA fragments with ds_load_b128.
// ---------------------------------------------------------------------------
__global__ __launch_bounds__(256) void gemm_lse_kernel(
    const u16* __restrict__ Abf,      // [8192][256] zs bf16
    const u16* __restrict__ Bbf,      // [32000][256] vocab bf16
    const float* __restrict__ bias,   // [32000]
    float* __restrict__ pmax,         // [8192][NCHUNKS]
    float* __restrict__ psum)         // [8192][NCHUNKS]
{
    __shared__ u16  s_a[MTILE * EMBED];   // 32 KB A tile
    __shared__ float lmax[MTILE][2];
    __shared__ float lsum[MTILE][2];

    const int mblk   = blockIdx.x / NCHUNKS;
    const int nchunk = blockIdx.x % NCHUNKS;
    const int mrow0  = mblk * MTILE;

    // ---- stage A tile (64x256 bf16 = 2048 x 16B) into LDS asynchronously ----
    {
        const unsigned s_a_base = (unsigned)(uintptr_t)&s_a[0];
        for (int c = threadIdx.x; c < (MTILE * EMBED) / 8; c += 256) {
            const int row  = c >> 5;          // 32 chunks of 8 bf16 per row
            const int col  = (c & 31) * 8;
            const u16* g   = Abf + (size_t)(mrow0 + row) * EMBED + col;
            const unsigned lds = s_a_base + (unsigned)c * 16u;
            asm volatile("global_load_async_to_lds_b128 %0, %1, off"
                         :: "v"(lds), "v"(g) : "memory");
        }
        asm volatile("s_wait_asynccnt 0x0" ::: "memory");
    }
    __syncthreads();

    const int lane = threadIdx.x & 31;
    const int wave = threadIdx.x >> 5;
    const int mt   = wave >> 1;        // 0..3: which 16-row subtile
    const int ng   = wave & 1;         // 0..1: which 80-col group
    const int hi   = lane >> 4;        // 0: lanes 0-15, 1: lanes 16-31
    const int l15  = lane & 15;
    const int n0   = nchunk * CHUNK_N + ng * 80;

    const u16* s_arow = s_a + (size_t)(mt * 16 + l15) * EMBED;

    v8f acc[5];
    #pragma unroll
    for (int a = 0; a < 5; ++a) acc[a] = v8f{0.f, 0.f, 0.f, 0.f, 0.f, 0.f, 0.f, 0.f};

    for (int k0 = 0; k0 < EMBED; k0 += 32) {
        const int ka = k0 + hi * 8;
        V16 af;                                                   // ds_load_b128 x2
        af.q[0] = *reinterpret_cast<const uint4*>(s_arow + ka);        // K ka..ka+7
        af.q[1] = *reinterpret_cast<const uint4*>(s_arow + ka + 16);   // K ka+16..+23
        #pragma unroll
        for (int a = 0; a < 5; ++a) {
            const u16* brow = Bbf + (size_t)(n0 + a * 16 + l15) * EMBED + k0 + hi * 8;
            V16 bf;
            bf.q[0] = *reinterpret_cast<const uint4*>(brow);      // K kb..kb+7, kb+16..kb+23 (kb = k0+8*hi)
            bf.q[1] = *reinterpret_cast<const uint4*>(brow + 16);
            if (k0 + 32 < EMBED)
                __builtin_prefetch(brow + 32, 0, 0);              // global_prefetch_b8
            acc[a] = __builtin_amdgcn_wmma_f32_16x16x32_bf16(
                false, af.v, false, bf.v, (short)0, acc[a], false, false);
        }
    }

    // add bias (per column, column = lane&15 within each acc set)
    #pragma unroll
    for (int a = 0; a < 5; ++a) {
        const float bv = bias[n0 + a * 16 + l15];
        #pragma unroll
        for (int r = 0; r < 8; ++r) acc[a][r] += bv;
    }

    // per-row (row = mt*16 + r + 8*hi) max and sum-exp over this wave's 80 cols
    #pragma unroll
    for (int r = 0; r < 8; ++r) {
        float rm = acc[0][r];
        #pragma unroll
        for (int a = 1; a < 5; ++a) rm = fmaxf(rm, acc[a][r]);
        #pragma unroll
        for (int mask = 1; mask <= 8; mask <<= 1)
            rm = fmaxf(rm, __shfl_xor(rm, mask, 32));
        float s = 0.f;
        #pragma unroll
        for (int a = 0; a < 5; ++a) s += expf(acc[a][r] - rm);
        #pragma unroll
        for (int mask = 1; mask <= 8; mask <<= 1)
            s += __shfl_xor(s, mask, 32);
        if (l15 == 0) {
            lmax[mt * 16 + r + hi * 8][ng] = rm;
            lsum[mt * 16 + r + hi * 8][ng] = s;
        }
    }
    __syncthreads();

    // merge the 2 column-groups -> one (max, sumexp) per row per 160-col chunk
    if (threadIdx.x < MTILE) {
        const int r = threadIdx.x;
        const float ma = lmax[r][0], mb = lmax[r][1];
        const float gm = fmaxf(ma, mb);
        const float gs = lsum[r][0] * expf(ma - gm) + lsum[r][1] * expf(mb - gm);
        const size_t o = (size_t)(mrow0 + r) * NCHUNKS + nchunk;
        pmax[o] = gm;
        psum[o] = gs;
    }
}

// ---------------------------------------------------------------------------
// Kernel 4: one wave per row. Exact f32 target dot + chunk logsumexp merge.
// yp = tgt_logit - (gmax + log(gsum))
// ---------------------------------------------------------------------------
__global__ __launch_bounds__(256) void finish_kernel(
    const int* __restrict__ y, const float* __restrict__ zs_f32,
    const float* __restrict__ vocab_w, const float* __restrict__ vocab_b,
    const float* __restrict__ pmax, const float* __restrict__ psum,
    float* __restrict__ out)
{
    const int row  = (blockIdx.x * blockDim.x + threadIdx.x) >> 5;
    const int lane = threadIdx.x & 31;
    if (row >= MROWS) return;

    const int v = y[row];
    const float* zr = zs_f32 + (size_t)row * EMBED;
    const float* wr = vocab_w + (size_t)v * EMBED;
    float dot = 0.f;
    #pragma unroll
    for (int e = lane; e < EMBED; e += 32) dot += zr[e] * wr[e];
    #pragma unroll
    for (int mask = 1; mask <= 16; mask <<= 1) dot += __shfl_xor(dot, mask, 32);
    const float tgt = dot + vocab_b[v];

    float m = -1e30f, s = 0.f;
    for (int c = lane; c < NCHUNKS; c += 32) {
        const float cm = pmax[(size_t)row * NCHUNKS + c];
        const float cs = psum[(size_t)row * NCHUNKS + c];
        const float nm = fmaxf(m, cm);
        s = s * expf(m - nm) + cs * expf(cm - nm);
        m = nm;
    }
    #pragma unroll
    for (int mask = 1; mask <= 16; mask <<= 1) {
        const float om = __shfl_xor(m, mask, 32);
        const float os = __shfl_xor(s, mask, 32);
        const float nm = fmaxf(m, om);
        s = s * expf(m - nm) + os * expf(om - nm);
        m = nm;
    }
    if (lane == 0) out[row] = tgt - (m + logf(s));
}

// ---------------------------------------------------------------------------
extern "C" void kernel_launch(void* const* d_in, const int* in_sizes, int n_in,
                              void* d_out, int out_size, void* d_ws, size_t ws_size,
                              hipStream_t stream) {
    const int*   zi      = (const int*)  d_in[0];
    const int*   y       = (const int*)  d_in[1];
    const float* latent  = (const float*)d_in[2];
    const float* kv      = (const float*)d_in[3];
    const float* vocab_w = (const float*)d_in[4];
    const float* vocab_b = (const float*)d_in[5];
    float* out = (float*)d_out;

    char* ws = (char*)d_ws;
    float* zs_f32  = (float*)(ws);                       // 8192*256*4  =  8,388,608 B
    u16*   zs_bf16 = (u16*)  (ws + 8388608);             // 8192*256*2  =  4,194,304 B
    u16*   w_bf16  = (u16*)  (ws + 12582912);            // 32000*256*2 = 16,384,000 B
    float* pmax    = (float*)(ws + 28966912);            // 8192*200*4  =  6,553,600 B
    float* psum    = (float*)(ws + 35520512);            // 8192*200*4  =  6,553,600 B
    (void)in_sizes; (void)n_in; (void)out_size; (void)ws_size;

    rnn_scan_kernel<<<BATCH, 256, 0, stream>>>(zi, latent, kv, zs_f32, zs_bf16);
    cvt_bf16_kernel<<<2048, 256, 0, stream>>>(vocab_w, w_bf16, VOCAB * EMBED);
    gemm_lse_kernel<<<(MROWS / MTILE) * NCHUNKS, 256, 0, stream>>>(
        zs_bf16, w_bf16, vocab_b, pmax, psum);
    finish_kernel<<<MROWS / 8, 256, 0, stream>>>(
        y, zs_f32, vocab_w, vocab_b, pmax, psum, out);
}
